// BidirectionalAttention_3169685864467
// MI455X (gfx1250) — matrix operand order repaired and checked by reading, not compiled, not run on hardware
//
#include <hip/hip_runtime.h>
#include <hip/hip_bf16.h>
#include <stdint.h>

typedef __attribute__((ext_vector_type(16))) _Float16 v16h;
typedef __attribute__((ext_vector_type(8)))  _Float16 v8h;
typedef __attribute__((ext_vector_type(8)))  float    v8f;

// ---------------- helpers ----------------

__device__ __forceinline__ v8f v8f_zero() {
  v8f z;
#pragma unroll
  for (int i = 0; i < 8; ++i) z[i] = 0.0f;
  return z;
}

// A-matrix fragment: 16x32 f16, source row-major with leading dim ld.
// ISA A layout (16-bit 16x32): lane l holds row m=l&15; within v16h,
// elements 0..7  -> k = (l>=16 ? 8:0) + 0..7
// elements 8..15 -> k = 16 + (l>=16 ? 8:0) + 0..7
__device__ __forceinline__ v16h load_fragA(const _Float16* p, int ld) {
  const int lane = threadIdx.x & 31;
  const int row  = lane & 15;
  const int kb   = (lane >> 4) << 3;   // 0 or 8
  const _Float16* r = p + row * ld + kb;
  v8h lo = *(const v8h*)(r);
  v8h hi = *(const v8h*)(r + 16);
  v16h a;
#pragma unroll
  for (int i = 0; i < 8; ++i) { a[i] = lo[i]; a[i + 8] = hi[i]; }
  return a;
}

// B-matrix fragment: 32x16 f16, loaded from BT (row-major [n][k], leading dim ld).
// ISA B layout: lane l holds column n=l&15; same k order as A:
// elements 0..7 -> k = (l>=16 ? 8:0) + 0..7, elements 8..15 -> k = 16 + (l>=16 ? 8:0) + 0..7
__device__ __forceinline__ v16h load_fragB(const _Float16* bt, int ld) {
  const int lane = threadIdx.x & 31;
  const int n    = lane & 15;
  const int k0   = (lane >> 4) << 3;   // 0 or 8
  const _Float16* r = bt + n * ld + k0;
  v8h lo = *(const v8h*)(r);
  v8h hi = *(const v8h*)(r + 16);
  v16h b;
#pragma unroll
  for (int i = 0; i < 8; ++i) { b[i] = lo[i]; b[i + 8] = hi[i]; }
  return b;
}

// ---------------- conversion kernels ----------------

__global__ void cvt_f32_to_f16_k(const float* __restrict__ src,
                                 _Float16* __restrict__ dst, int n) {
  int i = blockIdx.x * 256 + threadIdx.x;
  if (i < n) dst[i] = (_Float16)src[i];
}

// W [1024][1024] f32  ->  WT [n][k] f16 (transposed)
__global__ void cvt_transpose_k(const float* __restrict__ w,
                                _Float16* __restrict__ wt) {
  int i = blockIdx.x * 256 + threadIdx.x;   // 0 .. 1048575, coalesced read
  int k = i >> 10, n = i & 1023;
  wt[(n << 10) + k] = (_Float16)w[i];
}

// ---------------- WMMA GEMM ----------------
// C[M][N] = A[M][K] * BT[N][K]^T + bias
// Block tile 128x64 (4 waves), wave tile 32x64: 12 b128 loads / 8 WMMAs per
// k-step, all loads hoisted ahead of the WMMA burst for load/compute overlap.
// mode 0: C -> out16 row-major f16
// mode 1: C -> out32 row-major f32 (final projection)
// mode 2: C -> out16 per-head transposed: out16[((b*16+h)*64+d)*1024 + s]
__global__ __launch_bounds__(128) void gemm_wmma_k(
    const _Float16* __restrict__ A, const _Float16* __restrict__ BT,
    const float* __restrict__ bias, _Float16* __restrict__ out16,
    float* __restrict__ out32, int M, int N, int K, int mode) {
  const int wave = threadIdx.x >> 5;
  const int lane = threadIdx.x & 31;
  const int m0 = blockIdx.x * 128 + wave * 32;  // 32 rows per wave
  const int n0 = blockIdx.y * 64;               // 64 cols per block

  v8f acc[2][4];
#pragma unroll
  for (int mi = 0; mi < 2; ++mi)
#pragma unroll
    for (int t = 0; t < 4; ++t) acc[mi][t] = v8f_zero();

  const _Float16* aRow0 = A + (size_t)m0 * K;
  const _Float16* aRow1 = A + (size_t)(m0 + 16) * K;
  const _Float16* bRow0 = BT + (size_t)n0 * K;

  for (int k = 0; k < K; k += 32) {
    __builtin_prefetch(aRow0 + k + 64, 0, 3);
    // hoist all 12 fragment loads ahead of the 8-WMMA burst
    v16h a0 = load_fragA(aRow0 + k, K);
    v16h a1 = load_fragA(aRow1 + k, K);
    v16h b0 = load_fragB(bRow0 + k, K);
    v16h b1 = load_fragB(bRow0 + (size_t)16 * K + k, K);
    v16h b2 = load_fragB(bRow0 + (size_t)32 * K + k, K);
    v16h b3 = load_fragB(bRow0 + (size_t)48 * K + k, K);
    acc[0][0] = __builtin_amdgcn_wmma_f32_16x16x32_f16(false, a0, false, b0,
                                                       (short)0, acc[0][0], false, false);
    acc[0][1] = __builtin_amdgcn_wmma_f32_16x16x32_f16(false, a0, false, b1,
                                                       (short)0, acc[0][1], false, false);
    acc[0][2] = __builtin_amdgcn_wmma_f32_16x16x32_f16(false, a0, false, b2,
                                                       (short)0, acc[0][2], false, false);
    acc[0][3] = __builtin_amdgcn_wmma_f32_16x16x32_f16(false, a0, false, b3,
                                                       (short)0, acc[0][3], false, false);
    acc[1][0] = __builtin_amdgcn_wmma_f32_16x16x32_f16(false, a1, false, b0,
                                                       (short)0, acc[1][0], false, false);
    acc[1][1] = __builtin_amdgcn_wmma_f32_16x16x32_f16(false, a1, false, b1,
                                                       (short)0, acc[1][1], false, false);
    acc[1][2] = __builtin_amdgcn_wmma_f32_16x16x32_f16(false, a1, false, b2,
                                                       (short)0, acc[1][2], false, false);
    acc[1][3] = __builtin_amdgcn_wmma_f32_16x16x32_f16(false, a1, false, b3,
                                                       (short)0, acc[1][3], false, false);
  }

  const int mh = (lane >> 4) << 3;   // 0 or 8
  const int nn = lane & 15;
#pragma unroll
  for (int mi = 0; mi < 2; ++mi) {
#pragma unroll
    for (int t = 0; t < 4; ++t) {
      const int col = n0 + t * 16 + nn;
      const float bv = bias[col];
#pragma unroll
      for (int r = 0; r < 8; ++r) {
        const int row = m0 + mi * 16 + mh + r;
        const float val = acc[mi][t][r] + bv;
        if (mode == 1) {
          out32[(size_t)row * N + col] = val;
        } else if (mode == 0) {
          out16[(size_t)row * N + col] = (_Float16)val;
        } else {  // mode 2: V stored [b][h][d][s]
          const int b = row >> 10, s = row & 1023;
          const int h = col >> 6,  d = col & 63;
          out16[(size_t)(((b * 16 + h) * 64 + d) << 10) + s] = (_Float16)val;
        }
      }
    }
  }
}

// ---------------- flash attention ----------------
// Q,K: f16 [4096][1024] (cols = h*64+d).  Vt: f16 [b][h][64][1024].
// outA: f16 [4096][1024].  grid = (B*NH, S/64), block = 128 (4 waves).
__global__ __launch_bounds__(128) void attn_wmma_k(
    const _Float16* __restrict__ Q, const _Float16* __restrict__ Km,
    const _Float16* __restrict__ Vt, const unsigned char* __restrict__ mask,
    _Float16* __restrict__ outA) {
  constexpr int S = 1024, HTOT = 1024, HD = 64;
  const int bh = blockIdx.x;           // 0..63
  const int b = bh >> 4, h = bh & 15;
  const int wave = threadIdx.x >> 5;
  const int lane = threadIdx.x & 31;
  const int lh = lane >> 4;            // lane half
  const int ln = lane & 15;
  const int q0 = blockIdx.y * 64 + wave * 16;
  const float scale = 0.125f;          // 1/sqrt(64)

  __shared__ __align__(16) _Float16 ldsP[4][16 * 32];
  _Float16* myP = ldsP[wave];

  const _Float16* qBase = Q + (size_t)(b * S + q0) * HTOT + h * HD;
  const v16h qa0 = load_fragA(qBase, HTOT);
  const v16h qa1 = load_fragA(qBase + 32, HTOT);

  v8f oacc[4];
#pragma unroll
  for (int t = 0; t < 4; ++t) oacc[t] = v8f_zero();
  float mrow[8], lrow[8];
#pragma unroll
  for (int r = 0; r < 8; ++r) { mrow[r] = -1e30f; lrow[r] = 0.0f; }

  const _Float16* vBase = Vt + (size_t)bh * HD * S;

  for (int kb = 0; kb < S; kb += 32) {
    // --- scores: two 16x16 tiles covering keys kb..kb+31 ---
    const _Float16* kBase0 = Km + (size_t)(b * S + kb) * HTOT + h * HD;
    const _Float16* kBase1 = kBase0 + (size_t)16 * HTOT;
    v16h k00 = load_fragB(kBase0, HTOT);
    v16h k01 = load_fragB(kBase0 + 32, HTOT);
    v16h k10 = load_fragB(kBase1, HTOT);
    v16h k11 = load_fragB(kBase1 + 32, HTOT);
    v8f s[2];
    s[0] = v8f_zero();
    s[0] = __builtin_amdgcn_wmma_f32_16x16x32_f16(false, qa0, false, k00,
                                                  (short)0, s[0], false, false);
    s[0] = __builtin_amdgcn_wmma_f32_16x16x32_f16(false, qa1, false, k01,
                                                  (short)0, s[0], false, false);
    s[1] = v8f_zero();
    s[1] = __builtin_amdgcn_wmma_f32_16x16x32_f16(false, qa0, false, k10,
                                                  (short)0, s[1], false, false);
    s[1] = __builtin_amdgcn_wmma_f32_16x16x32_f16(false, qa1, false, k11,
                                                  (short)0, s[1], false, false);
#pragma unroll
    for (int t = 0; t < 2; ++t) {
      const int key = kb + t * 16 + ln;
      const bool mk = mask[b * S + key] != 0;
#pragma unroll
      for (int r = 0; r < 8; ++r) s[t][r] = mk ? -1e30f : s[t][r] * scale;
    }

    // --- online softmax (rows live across 16-lane halves) ---
    float corr[8], p0v[8], p1v[8];
#pragma unroll
    for (int r = 0; r < 8; ++r) {
      float mx = fmaxf(s[0][r], s[1][r]);
#pragma unroll
      for (int off = 8; off >= 1; off >>= 1)
        mx = fmaxf(mx, __shfl_xor(mx, off, 32));
      const float mnew = fmaxf(mrow[r], mx);
      const float c = __expf(mrow[r] - mnew);
      const float p0 = __expf(s[0][r] - mnew);
      const float p1 = __expf(s[1][r] - mnew);
      float rs = p0 + p1;
#pragma unroll
      for (int off = 8; off >= 1; off >>= 1) rs += __shfl_xor(rs, off, 32);
      lrow[r] = lrow[r] * c + rs;
      mrow[r] = mnew;
      corr[r] = c;
      p0v[r] = p0;
      p1v[r] = p1;
    }

    // --- re-layout probs (C-layout f32 -> A-layout f16) through LDS ---
#pragma unroll
    for (int r = 0; r < 8; ++r) {
      myP[(r + 8 * lh) * 32 + ln]      = (_Float16)p0v[r];
      myP[(r + 8 * lh) * 32 + 16 + ln] = (_Float16)p1v[r];
    }
    asm volatile("s_wait_dscnt 0" ::: "memory");
    const v16h pa = load_fragA(myP, 32);

    // --- O = diag(corr)*O + P * V ---
    v16h vb0 = load_fragB(vBase + (size_t)0  * S + kb, S);
    v16h vb1 = load_fragB(vBase + (size_t)16 * S + kb, S);
    v16h vb2 = load_fragB(vBase + (size_t)32 * S + kb, S);
    v16h vb3 = load_fragB(vBase + (size_t)48 * S + kb, S);
#pragma unroll
    for (int nt = 0; nt < 4; ++nt)
#pragma unroll
      for (int r = 0; r < 8; ++r) oacc[nt][r] *= corr[r];
    oacc[0] = __builtin_amdgcn_wmma_f32_16x16x32_f16(false, pa, false, vb0,
                                                     (short)0, oacc[0], false, false);
    oacc[1] = __builtin_amdgcn_wmma_f32_16x16x32_f16(false, pa, false, vb1,
                                                     (short)0, oacc[1], false, false);
    oacc[2] = __builtin_amdgcn_wmma_f32_16x16x32_f16(false, pa, false, vb2,
                                                     (short)0, oacc[2], false, false);
    oacc[3] = __builtin_amdgcn_wmma_f32_16x16x32_f16(false, pa, false, vb3,
                                                     (short)0, oacc[3], false, false);
    asm volatile("s_wait_dscnt 0" ::: "memory");  // LDS reads done before next iter's stores
  }

  // --- normalize and store f16 attn output [4096][1024] ---
#pragma unroll
  for (int nt = 0; nt < 4; ++nt) {
#pragma unroll
    for (int r = 0; r < 8; ++r) {
      const float val = oacc[nt][r] / lrow[r];
      const int row = q0 + r + 8 * lh;
      const int col = h * HD + nt * 16 + ln;
      outA[(size_t)(b * S + row) * HTOT + col] = (_Float16)val;
    }
  }
}

// ---------------- host launch ----------------

extern "C" void kernel_launch(void* const* d_in, const int* in_sizes, int n_in,
                              void* d_out, int out_size, void* d_ws, size_t ws_size,
                              hipStream_t stream) {
  const float*         x    = (const float*)d_in[0];
  const unsigned char* mask = (const unsigned char*)d_in[1];
  const float* wq = (const float*)d_in[2];
  const float* bq = (const float*)d_in[3];
  const float* wk = (const float*)d_in[4];
  const float* bk = (const float*)d_in[5];
  const float* wv = (const float*)d_in[6];
  const float* bv = (const float*)d_in[7];
  const float* wo = (const float*)d_in[8];
  const float* bo = (const float*)d_in[9];
  float* out = (float*)d_out;

  constexpr int M = 4096, N = 1024, K = 1024;
  constexpr size_t EM = (size_t)M * N;       // 4M elems
  constexpr size_t EW = (size_t)K * N;       // 1M elems

  _Float16* ws16 = (_Float16*)d_ws;
  _Float16* x16    = ws16;                   // 4M
  _Float16* wqT    = x16 + EM;               // 1M
  _Float16* wkT    = wqT + EW;
  _Float16* wvT    = wkT + EW;
  _Float16* woT    = wvT + EW;
  _Float16* Q16    = woT + EW;               // 4M
  _Float16* K16    = Q16 + EM;               // 4M
  _Float16* Vt16   = K16 + EM;               // 4M
  _Float16* attn16 = x16;                    // reuse x16 region

  cvt_f32_to_f16_k<<<(int)((EM + 255) / 256), 256, 0, stream>>>(x, x16, (int)EM);
  cvt_transpose_k<<<(int)(EW / 256), 256, 0, stream>>>(wq, wqT);
  cvt_transpose_k<<<(int)(EW / 256), 256, 0, stream>>>(wk, wkT);
  cvt_transpose_k<<<(int)(EW / 256), 256, 0, stream>>>(wv, wvT);
  cvt_transpose_k<<<(int)(EW / 256), 256, 0, stream>>>(wo, woT);

  dim3 gg(M / 128, N / 64), gb(128);
  gemm_wmma_k<<<gg, gb, 0, stream>>>(x16, wqT, bq, Q16, nullptr, M, N, K, 0);
  gemm_wmma_k<<<gg, gb, 0, stream>>>(x16, wkT, bk, K16, nullptr, M, N, K, 0);
  gemm_wmma_k<<<gg, gb, 0, stream>>>(x16, wvT, bv, Vt16, nullptr, M, N, K, 2);

  attn_wmma_k<<<dim3(64, 16), 128, 0, stream>>>(Q16, K16, Vt16, mask, attn16);

  gemm_wmma_k<<<gg, gb, 0, stream>>>(attn16, woT, bo, nullptr, out, M, N, K, 1);
}
